// HCAM_62466004353028
// MI455X (gfx1250) — matrix operand order repaired and checked by reading, not compiled, not run on hardware
//
#include <hip/hip_runtime.h>
#include <hip/hip_bf16.h>

typedef __attribute__((ext_vector_type(8)))  float  v8f;
typedef __attribute__((ext_vector_type(8)))  __bf16 v8bf;
typedef __attribute__((ext_vector_type(16))) __bf16 v16bf;

#define KP   36      // image patches per image
#define LW   50      // words per caption
#define DIM  1024
#define NKL  (KP * LW)   // 1800 elements per sparsemax

// ---- float -> bf16, round-to-nearest-even ----
static __device__ __forceinline__ unsigned short f2bf(float f) {
    unsigned int u = __float_as_uint(f);
    u += 0x7FFFu + ((u >> 16) & 1u);
    return (unsigned short)(u >> 16);
}

// ---------------------------------------------------------------------------
// Kernel 1/2: add EPS, L2-normalize each 1024-d row, emit bf16.
// 8 waves per block, one wave per row.
// ---------------------------------------------------------------------------
__global__ __launch_bounds__(256)
void hcam_norm_rows(const float* __restrict__ in,
                    unsigned short* __restrict__ out, int nrows) {
    const int row  = blockIdx.x * 8 + (threadIdx.x >> 5);
    const int lane = threadIdx.x & 31;
    if (row >= nrows) return;                      // uniform per wave

    const float* p = in + (size_t)row * DIM;
    float x[32];
    float ss = 0.f;
#pragma unroll
    for (int j = 0; j < 32; ++j) {
        float v = p[j * 32 + lane] + 1e-6f;
        x[j] = v;
        ss = fmaf(v, v, ss);
    }
#pragma unroll
    for (int o = 16; o > 0; o >>= 1) ss += __shfl_xor(ss, o, 32);
    const float inv = rsqrtf(ss);

    unsigned short* q = out + (size_t)row * DIM;
#pragma unroll
    for (int j = 0; j < 32; ++j) q[j * 32 + lane] = f2bf(x[j] * inv);
}

// A-fragment loader: lane (m, h=lane>>4): VGPR0-3 = K[8h .. 8h+7],
// VGPR4-7 = K[16+8h .. 16+8h+7]  (16-bit A 16x32 layout, wave32)
static __device__ __forceinline__ v16bf load_a_frag(const unsigned short* p) {
    v8bf lo = *(const v8bf*)(p);        // p already offset by 8*h
    v8bf hi = *(const v8bf*)(p + 16);
    return __builtin_shufflevector(lo, hi, 0, 1, 2, 3, 4, 5, 6, 7,
                                            8, 9, 10, 11, 12, 13, 14, 15);
}

// ---------------------------------------------------------------------------
// Kernel 3: one block per (i,t) pair. 4 waves:
//   wave w -> N-tile w (cols w*16..w*16+15), M-tiles 0..2 (rows 0..47, pad 36->48)
//   K loop: 1024 / 32 steps of v_wmma_f32_16x16x32_bf16
// Then masked 36x50 block -> LDS, sparsemax via bisection, weighted sum.
// ---------------------------------------------------------------------------
__global__ __launch_bounds__(128)
void hcam_block_kernel(const unsigned short* __restrict__ Abf,
                       const unsigned short* __restrict__ Bbf,
                       const int* __restrict__ img_lens,
                       const int* __restrict__ cap_lens,
                       float* __restrict__ out, int bt) {
    __shared__ float fg[NKL];
    __shared__ float red[4];

    const int tid  = threadIdx.x;
    const int lane = tid & 31;
    const int wv   = tid >> 5;      // 0..3  (N-tile)
    const int nn   = lane & 15;     // col-in-tile / row-in-tile
    const int h    = lane >> 4;     // K-half select
    const int i    = blockIdx.y;
    const int t    = blockIdx.x;

    const unsigned short* Ab = Abf + (size_t)i * KP * DIM;
    const unsigned short* Bb = Bbf + (size_t)t * LW * DIM;

    // B-fragment: lane (n, h) holds B_row[col][k0 + 8h .. +7] and [k0 + 16 + 8h .. +7]
    int col = wv * 16 + nn; if (col > LW - 1) col = LW - 1;   // clamp pad lanes
    const unsigned short* bp = Bb + (size_t)col * DIM + 8 * h;

    int r0 = nn;        if (r0 > KP - 1) r0 = KP - 1;
    int r1 = 16 + nn;   if (r1 > KP - 1) r1 = KP - 1;
    int r2 = 32 + nn;   if (r2 > KP - 1) r2 = KP - 1;
    const unsigned short* ap0 = Ab + (size_t)r0 * DIM + 8 * h;
    const unsigned short* ap1 = Ab + (size_t)r1 * DIM + 8 * h;
    const unsigned short* ap2 = Ab + (size_t)r2 * DIM + 8 * h;

    v8f acc0 = {}; v8f acc1 = {}; v8f acc2 = {};
#pragma unroll 4
    for (int k0 = 0; k0 < DIM; k0 += 32) {
        v16bf b  = load_a_frag(bp + k0);
        v16bf a0 = load_a_frag(ap0 + k0);
        v16bf a1 = load_a_frag(ap1 + k0);
        v16bf a2 = load_a_frag(ap2 + k0);
        acc0 = __builtin_amdgcn_wmma_f32_16x16x32_bf16(false, a0, false, b,
                                                       (short)0, acc0, false, false);
        acc1 = __builtin_amdgcn_wmma_f32_16x16x32_bf16(false, a1, false, b,
                                                       (short)0, acc1, false, false);
        acc2 = __builtin_amdgcn_wmma_f32_16x16x32_bf16(false, a2, false, b,
                                                       (short)0, acc2, false, false);
    }

    // Scatter masked C block into LDS. C layout: VGPR r, lane -> (M=r+8h, N=lane&15)
    const int ilen = img_lens[i];
    const int clen = cap_lens[t];
    const int N = wv * 16 + nn;
#pragma unroll
    for (int r = 0; r < 8; ++r) {
        int M0 = r + 8 * h;
        if (M0 < KP && N < LW) {
            float v = acc0[r]; if (M0 >= ilen || N >= clen) v = 0.f;
            fg[M0 * LW + N] = v;
        }
        int M1 = 16 + r + 8 * h;
        if (M1 < KP && N < LW) {
            float v = acc1[r]; if (M1 >= ilen || N >= clen) v = 0.f;
            fg[M1 * LW + N] = v;
        }
        int M2 = 32 + r + 8 * h;
        if (M2 < KP && N < LW) {
            float v = acc2[r]; if (M2 >= ilen || N >= clen) v = 0.f;
            fg[M2 * LW + N] = v;
        }
    }
    __syncthreads();

    // -------- sparsemax over 1800 values: bisection on tau --------
    float z[15];
#pragma unroll
    for (int j = 0; j < 15; ++j) {
        int idx = tid + j * 128;
        z[j] = (idx < NKL) ? fg[idx] : -1.0e30f;
    }

    // block-wide max
    float m = -1.0e30f;
#pragma unroll
    for (int j = 0; j < 15; ++j) m = fmaxf(m, z[j]);
#pragma unroll
    for (int o = 16; o > 0; o >>= 1) m = fmaxf(m, __shfl_xor(m, o, 32));
    if (lane == 0) red[wv] = m;
    __syncthreads();
    m = fmaxf(fmaxf(red[0], red[1]), fmaxf(red[2], red[3]));
    __syncthreads();

    // tau in (max-1, max]: f(tau) = sum(max(z-tau,0)) - 1 is decreasing
    float lo = m - 1.0f, hi = m;
    for (int it = 0; it < 40; ++it) {
        float tau = 0.5f * (lo + hi);
        float s = 0.f;
#pragma unroll
        for (int j = 0; j < 15; ++j) s += fmaxf(z[j] - tau, 0.f);
#pragma unroll
        for (int o = 16; o > 0; o >>= 1) s += __shfl_xor(s, o, 32);
        if (lane == 0) red[wv] = s;
        __syncthreads();
        s = red[0] + red[1] + red[2] + red[3];
        __syncthreads();
        if (s >= 1.0f) lo = tau; else hi = tau;
    }
    const float tau = 0.5f * (lo + hi);

    // result = sum(z * max(z - tau, 0))
    float a = 0.f;
#pragma unroll
    for (int j = 0; j < 15; ++j) a += z[j] * fmaxf(z[j] - tau, 0.f);
#pragma unroll
    for (int o = 16; o > 0; o >>= 1) a += __shfl_xor(a, o, 32);
    if (lane == 0) red[wv] = a;
    __syncthreads();
    if (tid == 0) out[(size_t)i * bt + t] = red[0] + red[1] + red[2] + red[3];
}

// ---------------------------------------------------------------------------
extern "C" void kernel_launch(void* const* d_in, const int* in_sizes, int n_in,
                              void* d_out, int out_size, void* d_ws, size_t ws_size,
                              hipStream_t stream) {
    // setup_inputs order: img_cls(0), imgs(1), cap_cls(2), caps(3),
    //                     img_lens(4), cap_lens(5)   (cls tensors unused)
    const float* imgs     = (const float*)d_in[1];
    const float* caps     = (const float*)d_in[3];
    const int*   img_lens = (const int*)d_in[4];
    const int*   cap_lens = (const int*)d_in[5];
    float*       out      = (float*)d_out;

    const int bi = in_sizes[4];
    const int bt = in_sizes[5];
    const int nA = bi * KP;     // 4608 rows
    const int nB = bt * LW;     // 6400 rows

    unsigned short* Abf = (unsigned short*)d_ws;                     // bi*36*1024 bf16
    unsigned short* Bbf = Abf + (size_t)nA * DIM;                    // bt*50*1024 bf16

    hcam_norm_rows<<<(nA + 7) / 8, 256, 0, stream>>>(imgs, Abf, nA);
    hcam_norm_rows<<<(nB + 7) / 8, 256, 0, stream>>>(caps, Bbf, nB);

    hcam_block_kernel<<<dim3(bt, bi), 128, 0, stream>>>(
        Abf, Bbf, img_lens, cap_lens, out, bt);
}
